// SimplicialConvLayer_13134009991681
// MI455X (gfx1250) — matrix operand order repaired and checked by reading, not compiled, not run on hardware
//
#include <hip/hip_runtime.h>
#include <hip/hip_bf16.h>

// ---------------------------------------------------------------------------
// CDNA5 (gfx1250) implementation of the simplicial conv layer.
// Heavy GEMMs run on the WMMA matrix cores (v_wmma_f32_16x16x32_bf16),
// converting fp32 -> bf16 on the fly while staging tiles into LDS.
// Pipelined: register prefetch of the next K-tile overlaps the WMMAs,
// LDS double-buffered, one workgroup barrier per K-step.
// ---------------------------------------------------------------------------

typedef __attribute__((ext_vector_type(16))) __bf16        bf16x16;
typedef __attribute__((ext_vector_type(8)))  float         floatx8;
typedef __attribute__((ext_vector_type(8)))  unsigned int  uintx8;

__device__ __forceinline__ unsigned short f2b(float f) {
  // round-to-nearest-even f32 -> bf16
  unsigned int u = __float_as_uint(f);
  u += 0x7FFFu + ((u >> 16) & 1u);
  return (unsigned short)(u >> 16);
}

__device__ __forceinline__ bf16x16 ld_frag(const unsigned short* p0,
                                           const unsigned short* p1) {
  const uint4 lo = *(const uint4*)p0;   // ds_load_b128
  const uint4 hi = *(const uint4*)p1;   // ds_load_b128
  uintx8 u = { lo.x, lo.y, lo.z, lo.w, hi.x, hi.y, hi.z, hi.w };
  return __builtin_bit_cast(bf16x16, u);
}

// branch-free tanh: t = 1 - 2/(exp(2|x|)+1), sign restored.
// exp via v_exp_f32 (exp2), reciprocal via v_rcp_f32; exact limit t->1 as |x|->inf.
__device__ __forceinline__ float fast_tanh(float x) {
  const float ax = fabsf(x);
  const float e2 = __builtin_amdgcn_exp2f(ax * 2.8853900817779268f); // 2*log2(e)
  const float t  = 1.f - 2.f * __builtin_amdgcn_rcpf(e2 + 1.f);
  return copysignf(t, x);
}

// ---------------------------------------------------------------------------
// Generic GEMM: C[M,N] = act( A[M,K] @ B[K,N] + bias[N] )
// A, B, C fp32 row-major; converted to bf16 during LDS staging.
// Tile 128x128, K-step 32, 256 threads (8 wave32), wave grid 4(M) x 2(N).
// Requires M%128==0, N%128==0, K%32==0 (true for every shape used here).
// ---------------------------------------------------------------------------
#define TM 128
#define TN 128
#define TK 32
#define LDSA 40   // padded LDS row stride in bf16 elements (80B, 16B aligned)

template <int BIAS, int ACT>
__global__ __launch_bounds__(256) void k_gemm_bf16wmma(
    const float* __restrict__ A, const float* __restrict__ B,
    const float* __restrict__ bias, float* __restrict__ C,
    int M, int N, int K)
{
  __shared__ alignas(16) unsigned short sA[2][TM * LDSA]; // M x K tiles
  __shared__ alignas(16) unsigned short sB[2][TN * LDSA]; // N x K tiles (transposed)

  const int tid    = threadIdx.x;
  const int tileM0 = blockIdx.y * TM;
  const int tileN0 = blockIdx.x * TN;

  const int lane = tid & 31;
  const int wave = tid >> 5;
  const int wm   = wave & 3;    // 4 wave-rows, 32 M each
  const int wn   = wave >> 2;   // 2 wave-cols, 64 N each
  const int l16  = lane & 15;
  const int half = lane >> 4;   // lane-group within wave (ISA 16-bit layouts)

  floatx8 acc[2][4] = {};
  float4 pa[4], pb[4];          // register prefetch staging

  auto gload = [&](int k0) {
#pragma unroll
    for (int j = 0; j < 4; ++j) {
      const int q = tid + j * 256;          // 0..1023
      const int row = q >> 3, kq = q & 7;   // A: 8 float4 per M-row
      pa[j] = *(const float4*)(A + (size_t)(tileM0 + row) * K + k0 + kq * 4);
      const int kk = q >> 5, nq = q & 31;   // B: 32 float4 per K-row
      pb[j] = *(const float4*)(B + (size_t)(k0 + kk) * N + tileN0 + nq * 4);
    }
  };
  auto lstore = [&](int buf) {
#pragma unroll
    for (int j = 0; j < 4; ++j) {
      const int q = tid + j * 256;
      const int row = q >> 3, kq = q & 7;
      ushort4 h;
      h.x = f2b(pa[j].x); h.y = f2b(pa[j].y); h.z = f2b(pa[j].z); h.w = f2b(pa[j].w);
      *(ushort4*)&sA[buf][row * LDSA + kq * 4] = h;
      const int kk = q >> 5, nq = q & 31;   // transpose B to n-major
      sB[buf][(nq * 4 + 0) * LDSA + kk] = f2b(pb[j].x);
      sB[buf][(nq * 4 + 1) * LDSA + kk] = f2b(pb[j].y);
      sB[buf][(nq * 4 + 2) * LDSA + kk] = f2b(pb[j].z);
      sB[buf][(nq * 4 + 3) * LDSA + kk] = f2b(pb[j].w);
    }
  };

  // prologue: stage tile 0
  gload(0);
  lstore(0);
  __syncthreads();

  int cur = 0;
  for (int k0 = 0; k0 < K; k0 += TK) {
    const bool nxt = (k0 + TK) < K;
    if (nxt) gload(k0 + TK);    // global loads in flight during the WMMAs

    // ---- fragments per ISA 16-bit WMMA layouts ----
    bf16x16 af[2], bfr[4];
#pragma unroll
    for (int mi = 0; mi < 2; ++mi) {
      // A 16x32: lane -> row m=l16; elems 0..7 k=half*8+i, 8..15 k=16+half*8+(i-8)
      const int r = wm * 32 + mi * 16 + l16;
      af[mi] = ld_frag(&sA[cur][r * LDSA + half * 8],
                       &sA[cur][r * LDSA + 16 + half * 8]);
    }
#pragma unroll
    for (int ni = 0; ni < 4; ++ni) {
      // B 32x16: lane -> col n=l16; elems 0..7 k=half*8+i, 8..15 k=16+half*8+(i-8)
      const int n = wn * 64 + ni * 16 + l16;
      bfr[ni] = ld_frag(&sB[cur][n * LDSA + half * 8],
                        &sB[cur][n * LDSA + 16 + half * 8]);
    }

#pragma unroll
    for (int mi = 0; mi < 2; ++mi)
#pragma unroll
      for (int ni = 0; ni < 4; ++ni)
        acc[mi][ni] = __builtin_amdgcn_wmma_f32_16x16x32_bf16(
            false, af[mi], false, bfr[ni], (short)0, acc[mi][ni], false, false);

    if (nxt) lstore(cur ^ 1);   // fill the other buffer
    __syncthreads();            // one barrier per K-step
    cur ^= 1;
  }

  // ---- epilogue: C/D layout -> VGPR r holds M = half*8 + r, N = l16 ----
#pragma unroll
  for (int mi = 0; mi < 2; ++mi) {
    const int mbase = tileM0 + wm * 32 + mi * 16 + half * 8;
#pragma unroll
    for (int ni = 0; ni < 4; ++ni) {
      const int n = tileN0 + wn * 64 + ni * 16 + l16;
      const float bv = BIAS ? bias[n] : 0.f;
#pragma unroll
      for (int r = 0; r < 8; ++r) {
        float c = acc[mi][ni][r] + bv;
        if (ACT) c = fast_tanh(c);
        C[(size_t)(mbase + r) * N + n] = c;
      }
    }
  }
}

// ---------------------------------------------------------------------------
// Elementwise / scatter / gather helpers (fp32, bandwidth-bound)
// ---------------------------------------------------------------------------
__global__ void k_zero(float* __restrict__ p, size_t n) {
  size_t i = (size_t)blockIdx.x * blockDim.x + threadIdx.x;
  size_t stride = (size_t)gridDim.x * blockDim.x;
  for (; i < n; i += stride) p[i] = 0.f;
}

__global__ void k_count(const int* __restrict__ ids, int n, float* __restrict__ cnt) {
  int i = blockIdx.x * blockDim.x + threadIdx.x;
  if (i < n) atomicAdd(&cnt[ids[i]], 1.f);
}

// one block per (source-row, incidence-slot); acc[idx[pair]] += src[pair/arity]
__global__ void k_scatter_rows(const float* __restrict__ src,
                               const int* __restrict__ idx,
                               int arity, int H, float* __restrict__ acc) {
  const int pair = blockIdx.x;
  const int srow = pair / arity;
  const int drow = idx[pair];
  const float* s = src + (size_t)srow * H;
  float*       d = acc + (size_t)drow * H;
  for (int h = threadIdx.x; h < H; h += blockDim.x)
    atomicAdd(&d[h], s[h]);
}

__global__ void k_div_rows(float* __restrict__ acc, const float* __restrict__ cnt,
                           size_t total, int H) {
  size_t i = (size_t)blockIdx.x * blockDim.x + threadIdx.x;
  size_t stride = (size_t)gridDim.x * blockDim.x;
  for (; i < total; i += stride) {
    float c = cnt[i / H];
    acc[i] /= (c > 1.f ? c : 1.f);
  }
}

// eacc = eacc/max(tcnt,1) + 0.5*(nmsg[v0] + nmsg[v1])
__global__ void k_edge_combine(float* __restrict__ eacc, const float* __restrict__ tcnt,
                               const float* __restrict__ nmsg, const int* __restrict__ edges,
                               size_t total, int H) {
  size_t i = (size_t)blockIdx.x * blockDim.x + threadIdx.x;
  size_t stride = (size_t)gridDim.x * blockDim.x;
  for (; i < total; i += stride) {
    const size_t e = i / H;
    const int    h = (int)(i % H);
    const int v0 = edges[e * 2 + 0];
    const int v1 = edges[e * 2 + 1];
    float c = tcnt[e]; c = (c > 1.f ? c : 1.f);
    eacc[i] = eacc[i] / c +
              0.5f * (nmsg[(size_t)v0 * H + h] + nmsg[(size_t)v1 * H + h]);
  }
}

// out[t] = mean over 3 boundary edges of emsg
__global__ void k_gather_tri(const float* __restrict__ emsg, const int* __restrict__ tri_edges,
                             float* __restrict__ out, size_t total, int H) {
  size_t i = (size_t)blockIdx.x * blockDim.x + threadIdx.x;
  size_t stride = (size_t)gridDim.x * blockDim.x;
  for (; i < total; i += stride) {
    const size_t t = i / H;
    const int    h = (int)(i % H);
    const int e0 = tri_edges[t * 3 + 0];
    const int e1 = tri_edges[t * 3 + 1];
    const int e2 = tri_edges[t * 3 + 2];
    out[i] = (emsg[(size_t)e0 * H + h] + emsg[(size_t)e1 * H + h] +
              emsg[(size_t)e2 * H + h]) * (1.f / 3.f);
  }
}

__global__ void k_concat(const float* __restrict__ a, int D1,
                         const float* __restrict__ b, int D2,
                         float* __restrict__ out, size_t total) {
  const int D = D1 + D2;
  size_t i = (size_t)blockIdx.x * blockDim.x + threadIdx.x;
  size_t stride = (size_t)gridDim.x * blockDim.x;
  for (; i < total; i += stride) {
    const size_t r = i / D;
    const int    c = (int)(i % D);
    out[i] = (c < D1) ? a[r * D1 + c] : b[r * D2 + (c - D1)];
  }
}

__global__ void k_combine(const float* __restrict__ feat, const float* __restrict__ mlp,
                          const float* __restrict__ smooth, float* __restrict__ out, size_t n) {
  size_t i = (size_t)blockIdx.x * blockDim.x + threadIdx.x;
  size_t stride = (size_t)gridDim.x * blockDim.x;
  for (; i < n; i += stride)
    out[i] = feat[i] + 0.1f * mlp[i] - 0.05f * smooth[i];
}

// ---------------------------------------------------------------------------
// Host-side orchestration
// ---------------------------------------------------------------------------
extern "C" void kernel_launch(void* const* d_in, const int* in_sizes, int n_in,
                              void* d_out, int out_size, void* d_ws, size_t ws_size,
                              hipStream_t stream) {
  (void)in_sizes; (void)n_in; (void)out_size; (void)ws_size;

  const int Nn = 4096, Ee = 16384, Tt = 8192;
  const int FD = 256;   // ND == ED == TD == 256
  const int H  = 512;

  // Inputs in reference dict insertion order; params depth-first (w, b; l1, l2)
  const float* node_feat = (const float*)d_in[0];
  const float* edge_feat = (const float*)d_in[1];
  const float* tri_feat  = (const float*)d_in[2];
  const float* L0        = (const float*)d_in[3];
  const float* L1        = (const float*)d_in[4];
  const float* L2        = (const float*)d_in[5];
  const int*   edges     = (const int*)d_in[6];
  const int*   tri_edges = (const int*)d_in[7];
  const float* nu_w1 = (const float*)d_in[8];
  const float* nu_b1 = (const float*)d_in[9];
  const float* nu_w2 = (const float*)d_in[10];
  const float* nu_b2 = (const float*)d_in[11];
  const float* eu_w1 = (const float*)d_in[12];
  const float* eu_b1 = (const float*)d_in[13];
  const float* eu_w2 = (const float*)d_in[14];
  const float* eu_b2 = (const float*)d_in[15];
  const float* tu_w1 = (const float*)d_in[16];
  const float* tu_b1 = (const float*)d_in[17];
  const float* tu_w2 = (const float*)d_in[18];
  const float* tu_b2 = (const float*)d_in[19];
  const float* e2n_w = (const float*)d_in[20];
  const float* e2n_b = (const float*)d_in[21];
  const float* n2e_w = (const float*)d_in[22];
  const float* n2e_b = (const float*)d_in[23];
  const float* t2e_w = (const float*)d_in[24];
  const float* t2e_b = (const float*)d_in[25];
  const float* e2t_w = (const float*)d_in[26];
  const float* e2t_b = (const float*)d_in[27];

  // ---- workspace layout (floats); ~290 MB total ----
  float* ws = (float*)d_ws;
  size_t off = 0;
  auto wsa = [&](size_t n) { float* p = ws + off; off += n; return p; };
  float* nsm   = wsa((size_t)Nn * FD);      // node_smooth
  float* esm   = wsa((size_t)Ee * FD);      // edge_smooth
  float* tsm   = wsa((size_t)Tt * FD);      // tri_smooth
  float* nmsg  = wsa((size_t)Nn * H);       // node_to_edge output
  float* tmsg  = wsa((size_t)Tt * H);       // tri_to_edge output
  float* eacc  = wsa((size_t)Ee * H);       // tri->edge scatter, then combined edge msg
  float* nacc  = wsa((size_t)Nn * H);       // edge->node scatter (node_from_edges)
  float* tgat  = wsa((size_t)Tt * H);       // tri_from_edges
  float* deg   = wsa((size_t)Nn);
  float* tcnt  = wsa((size_t)Ee);
  float* scrA  = wsa((size_t)Ee * (FD + H)); // edge_msg -> edge_msg_tri -> edge_cat
  float* ncat  = wsa((size_t)Nn * (FD + H));
  float* tcat  = wsa((size_t)Tt * (FD + H));
  float* nh    = wsa((size_t)Nn * H);
  float* eh    = wsa((size_t)Ee * H);
  float* th    = wsa((size_t)Tt * H);
  float* nmlp  = wsa((size_t)Nn * FD);
  float* emlp  = wsa((size_t)Ee * FD);
  float* tmlp  = wsa((size_t)Tt * FD);

  float* out_node = (float*)d_out;
  float* out_edge = out_node + (size_t)Nn * FD;
  float* out_tri  = out_edge + (size_t)Ee * FD;

  auto gemm = [&](const float* A, const float* B, const float* bias, float* C,
                  int M, int Nc, int K, int act) {
    dim3 grid(Nc / TN, M / TM);
    if (bias) {
      if (act) k_gemm_bf16wmma<1, 1><<<grid, 256, 0, stream>>>(A, B, bias, C, M, Nc, K);
      else     k_gemm_bf16wmma<1, 0><<<grid, 256, 0, stream>>>(A, B, bias, C, M, Nc, K);
    } else {
      if (act) k_gemm_bf16wmma<0, 1><<<grid, 256, 0, stream>>>(A, B, bias, C, M, Nc, K);
      else     k_gemm_bf16wmma<0, 0><<<grid, 256, 0, stream>>>(A, B, bias, C, M, Nc, K);
    }
  };
  auto egrid = [](size_t n) {
    size_t b = (n + 255) / 256;
    return (unsigned)(b > 16384 ? 16384 : b);
  };

  // 0) zero atomic accumulation buffers (ws is poisoned, not re-zeroed)
  k_zero<<<egrid((size_t)Nn * H), 256, 0, stream>>>(nacc, (size_t)Nn * H);
  k_zero<<<egrid((size_t)Nn), 256, 0, stream>>>(deg, (size_t)Nn);
  k_zero<<<egrid((size_t)Ee * H), 256, 0, stream>>>(eacc, (size_t)Ee * H);
  k_zero<<<egrid((size_t)Ee), 256, 0, stream>>>(tcnt, (size_t)Ee);

  // 1) Laplacian smoothing GEMMs (the heavy hitters)
  gemm(L0, node_feat, nullptr, nsm, Nn, FD, Nn, 0);
  gemm(L1, edge_feat, nullptr, esm, Ee, FD, Ee, 0);
  gemm(L2, tri_feat,  nullptr, tsm, Tt, FD, Tt, 0);

  // 2) edges -> nodes: edge_msg = edge_feat @ W + b, then incidence-mean
  gemm(edge_feat, e2n_w, e2n_b, scrA, Ee, H, FD, 0);
  k_count<<<(2 * Ee + 255) / 256, 256, 0, stream>>>(edges, 2 * Ee, deg);
  k_scatter_rows<<<2 * Ee, 128, 0, stream>>>(scrA, edges, 2, H, nacc);
  k_div_rows<<<egrid((size_t)Nn * H), 256, 0, stream>>>(nacc, deg, (size_t)Nn * H, H);

  // 3) nodes -> edges and triangles -> edges
  gemm(node_feat, n2e_w, n2e_b, nmsg, Nn, H, FD, 0);
  gemm(tri_feat,  t2e_w, t2e_b, tmsg, Tt, H, FD, 0);
  k_count<<<(3 * Tt + 255) / 256, 256, 0, stream>>>(tri_edges, 3 * Tt, tcnt);
  k_scatter_rows<<<3 * Tt, 128, 0, stream>>>(tmsg, tri_edges, 3, H, eacc);
  k_edge_combine<<<egrid((size_t)Ee * H), 256, 0, stream>>>(
      eacc, tcnt, nmsg, edges, (size_t)Ee * H, H);

  // 4) edges -> triangles (reuse scrA for edge_msg_tri; prior use finished)
  gemm(edge_feat, e2t_w, e2t_b, scrA, Ee, H, FD, 0);
  k_gather_tri<<<egrid((size_t)Tt * H), 256, 0, stream>>>(
      scrA, tri_edges, tgat, (size_t)Tt * H, H);

  // 5) concat [feat | msg] inputs for the update MLPs (scrA reused as edge_cat)
  k_concat<<<egrid((size_t)Nn * (FD + H)), 256, 0, stream>>>(
      node_feat, FD, nacc, H, ncat, (size_t)Nn * (FD + H));
  k_concat<<<egrid((size_t)Ee * (FD + H)), 256, 0, stream>>>(
      edge_feat, FD, eacc, H, scrA, (size_t)Ee * (FD + H));
  k_concat<<<egrid((size_t)Tt * (FD + H)), 256, 0, stream>>>(
      tri_feat, FD, tgat, H, tcat, (size_t)Tt * (FD + H));

  // 6) update MLPs: l1 (+bias, tanh) then l2 (+bias)
  gemm(ncat, nu_w1, nu_b1, nh, Nn, H, FD + H, 1);
  gemm(scrA, eu_w1, eu_b1, eh, Ee, H, FD + H, 1);
  gemm(tcat, tu_w1, tu_b1, th, Tt, H, FD + H, 1);
  gemm(nh, nu_w2, nu_b2, nmlp, Nn, FD, H, 0);
  gemm(eh, eu_w2, eu_b2, emlp, Ee, FD, H, 0);
  gemm(th, tu_w2, tu_b2, tmlp, Tt, FD, H, 0);

  // 7) residual + smoothing combine -> outputs
  k_combine<<<egrid((size_t)Nn * FD), 256, 0, stream>>>(
      node_feat, nmlp, nsm, out_node, (size_t)Nn * FD);
  k_combine<<<egrid((size_t)Ee * FD), 256, 0, stream>>>(
      edge_feat, emlp, esm, out_edge, (size_t)Ee * FD);
  k_combine<<<egrid((size_t)Tt * FD), 256, 0, stream>>>(
      tri_feat, tmlp, tsm, out_tri, (size_t)Tt * FD);
}
